// My_lstm_50414326120716
// MI455X (gfx1250) — matrix operand order repaired and checked by reading, not compiled, not run on hardware
//
#include <hip/hip_runtime.h>
#include <hip/hip_bf16.h>
#include <stdint.h>

#define HIDN   1024
#define BATCH  256
#define TENC   128
#define PRED   96
#define OUTW   7

typedef __bf16 bf16_t;
typedef __attribute__((ext_vector_type(16))) __bf16 v16bf;
typedef __attribute__((ext_vector_type(8)))  __bf16 v8bf;
typedef __attribute__((ext_vector_type(8)))  float   v8f;

__device__ __forceinline__ float sigmoidf_(float x) {
    return 1.0f / (1.0f + __expf(-x));
}

// ---------------- prep kernels ----------------

__global__ void fill_u32_kernel(uint32_t* __restrict__ p, uint32_t v, int n) {
    int i = blockIdx.x * blockDim.x + threadIdx.x;
    if (i < n) p[i] = v;
}

__global__ void cvt_bf16_kernel(const float* __restrict__ s, bf16_t* __restrict__ d, int n) {
    int i = blockIdx.x * blockDim.x + threadIdx.x;
    if (i < n) d[i] = (bf16_t)s[i];
}

// xin[b,t,0:7]=x[b,t,:], xin[b,t,7:10]=sum of 4 embedding rows
__global__ void build_xin_kernel(const float* __restrict__ x, const int* __restrict__ xte,
                                 const float* __restrict__ e0, const float* __restrict__ e1,
                                 const float* __restrict__ e2, const float* __restrict__ e3,
                                 float* __restrict__ xin) {
    int i = blockIdx.x * blockDim.x + threadIdx.x;   // over B*T
    if (i >= BATCH * TENC) return;
    const int* idx = xte + (size_t)i * 4;
    int i0 = idx[0], i1 = idx[1], i2 = idx[2], i3 = idx[3];
    float* o = xin + (size_t)i * 10;
    const float* xr = x + (size_t)i * 7;
#pragma unroll
    for (int k = 0; k < 7; ++k) o[k] = xr[k];
#pragma unroll
    for (int c = 0; c < 3; ++c)
        o[7 + c] = e0[i0 * 3 + c] + e1[i1 * 3 + c] + e2[i2 * 3 + c] + e3[i3 * 3 + c];
}

// dte[b,d,0:3] = sum of 4 embedding rows for decoder time codes
__global__ void build_dte_kernel(const int* __restrict__ yte,
                                 const float* __restrict__ e0, const float* __restrict__ e1,
                                 const float* __restrict__ e2, const float* __restrict__ e3,
                                 float* __restrict__ dte) {
    int i = blockIdx.x * blockDim.x + threadIdx.x;   // over B*PRED
    if (i >= BATCH * PRED) return;
    const int* idx = yte + (size_t)i * 4;
    int i0 = idx[0], i1 = idx[1], i2 = idx[2], i3 = idx[3];
    float* o = dte + (size_t)i * 3;
#pragma unroll
    for (int c = 0; c < 3; ++c)
        o[c] = e0[i0 * 3 + c] + e1[i1 * 3 + c] + e2[i2 * 3 + c] + e3[i3 * 3 + c];
}

// ---------------- WMMA fragment loads ----------------
// A fragment 16x32 bf16 (M x K): lane<16 -> M=lane, K = {k0..k0+7, k0+16..k0+23}
//                                lane>=16 -> M=lane-16, K = {k0+8..k0+15, k0+24..k0+31}
__device__ __forceinline__ v16bf load_a_frag(const bf16_t* __restrict__ act, int lda,
                                             int m0, int k0, int lane) {
    int m  = m0 + (lane & 15);
    int kc = k0 + ((lane < 16) ? 0 : 8);
    const bf16_t* p = act + (size_t)m * lda + kc;
    v8bf lo = *(const v8bf*)p;          // K = kc .. kc+7
    v8bf hi = *(const v8bf*)(p + 16);   // K = kc+16 .. kc+23
    return __builtin_shufflevector(lo, hi, 0,1,2,3,4,5,6,7,8,9,10,11,12,13,14,15);
}

// B fragment 32x16 bf16 (K x N) where B[k][n] = W[n0+n][k0+k] (W row-major [rows][ldw]):
// lane<16 -> N=lane, K = {k0..k0+7, k0+16..k0+23} ; lane>=16 -> N=lane-16, K = {k0+8..k0+15, k0+24..k0+31}
__device__ __forceinline__ v16bf load_b_frag(const bf16_t* __restrict__ W, int ldw,
                                             int n0, int k0, int lane) {
    int n  = n0 + (lane & 15);
    int kb = k0 + ((lane < 16) ? 0 : 8);
    const bf16_t* p = W + (size_t)n * ldw + kb;
    v8bf lo = *(const v8bf*)p;
    v8bf hi = *(const v8bf*)(p + 16);
    return __builtin_shufflevector(lo, hi, 0,1,2,3,4,5,6,7,8,9,10,11,12,13,14,15);
}

// ---------------- fused gate GEMM + LSTM cell update ----------------
// Block = one j-tile (16 hidden cols, all 4 gates); its 8 waves cover the 8 M-tile
// pairs (32 batch rows each). All waves in a block issue IDENTICAL weight-fragment
// addresses -> B fragments cross L2->WGP once per block; weights stream from L2
// roughly once per time-step in aggregate.
__global__ __launch_bounds__(256)
void lstm_gate_kernel(const bf16_t* __restrict__ hprev,  // [B][H]  recurrent h(t-1), bf16
                      const bf16_t* __restrict__ Whh,    // [4H][H] bf16
                      const bf16_t* __restrict__ xh,     // [B][H]  layer-below h(t), bf16 (or null)
                      const bf16_t* __restrict__ Wih,    // [4H][H] bf16 (or null)
                      const float*  __restrict__ x0,     // layer-0 input rows of 10 floats (or null)
                      int x0_stride,                     // elements between batch rows of x0
                      const float*  __restrict__ Wih0,   // [4H][10] f32 (or null)
                      const float*  __restrict__ bih,    // [4H]
                      const float*  __restrict__ bhh,    // [4H]
                      float*        __restrict__ c,      // [B][H] f32, in-place
                      bf16_t*       __restrict__ hout)   // [B][H] bf16, h(t)
{
    int lane = threadIdx.x & 31;
    int mt = threadIdx.x >> 5;   // 0..7  -> 32 batch rows each (waves of this block)
    int jt = blockIdx.x;         // 0..63 -> 16 hidden cols each
    int m0 = mt * 32;
    int j0 = jt * 16;

    v8f zf = {};
    v8f acc[2][4];
#pragma unroll
    for (int mh = 0; mh < 2; ++mh)
#pragma unroll
        for (int g = 0; g < 4; ++g) acc[mh][g] = zf;

    // recurrent GEMM: gates += hprev @ Whh^T
    for (int k0 = 0; k0 < HIDN; k0 += 32) {
        v16bf a0 = load_a_frag(hprev, HIDN, m0,      k0, lane);
        v16bf a1 = load_a_frag(hprev, HIDN, m0 + 16, k0, lane);
#pragma unroll
        for (int g = 0; g < 4; ++g) {
            v16bf b = load_b_frag(Whh, HIDN, g * HIDN + j0, k0, lane);
            acc[0][g] = __builtin_amdgcn_wmma_f32_16x16x32_bf16(
                            false, a0, false, b, (short)0, acc[0][g], false, false);
            acc[1][g] = __builtin_amdgcn_wmma_f32_16x16x32_bf16(
                            false, a1, false, b, (short)0, acc[1][g], false, false);
        }
    }

    // input GEMM for layers 1,2: gates += xh @ Wih^T
    if (xh != nullptr) {
        for (int k0 = 0; k0 < HIDN; k0 += 32) {
            v16bf a0 = load_a_frag(xh, HIDN, m0,      k0, lane);
            v16bf a1 = load_a_frag(xh, HIDN, m0 + 16, k0, lane);
#pragma unroll
            for (int g = 0; g < 4; ++g) {
                v16bf b = load_b_frag(Wih, HIDN, g * HIDN + j0, k0, lane);
                acc[0][g] = __builtin_amdgcn_wmma_f32_16x16x32_bf16(
                                false, a0, false, b, (short)0, acc[0][g], false, false);
                acc[1][g] = __builtin_amdgcn_wmma_f32_16x16x32_bf16(
                                false, a1, false, b, (short)0, acc[1][g], false, false);
            }
        }
    }

    int n = lane & 15;                // this lane's column within the j-tile
    int jcol = j0 + n;

    // layer-0 10-wide input projection (VALU, fp32)
    if (x0 != nullptr) {
        float wrow[4][10];
#pragma unroll
        for (int g = 0; g < 4; ++g) {
            const float* wr = Wih0 + (size_t)(g * HIDN + jcol) * 10;
#pragma unroll
            for (int d = 0; d < 10; ++d) wrow[g][d] = wr[d];
        }
#pragma unroll
        for (int mh = 0; mh < 2; ++mh) {
#pragma unroll
            for (int e = 0; e < 8; ++e) {
                int brow = m0 + mh * 16 + e + ((lane < 16) ? 0 : 8);
                const float* xr = x0 + (size_t)brow * x0_stride;
                float xv[10];
#pragma unroll
                for (int d = 0; d < 10; ++d) xv[d] = xr[d];
#pragma unroll
                for (int g = 0; g < 4; ++g) {
                    float s = 0.f;
#pragma unroll
                    for (int d = 0; d < 10; ++d) s += xv[d] * wrow[g][d];
                    acc[mh][g][e] += s;
                }
            }
        }
    }

    // biases (i,f,g,o)
    float bsum[4];
#pragma unroll
    for (int g = 0; g < 4; ++g) {
        int col = g * HIDN + jcol;
        bsum[g] = bih[col] + bhh[col];
    }

    // fused cell update: c_new = sig(f)*c + sig(i)*tanh(g); h = sig(o)*tanh(c_new)
#pragma unroll
    for (int mh = 0; mh < 2; ++mh) {
#pragma unroll
        for (int e = 0; e < 8; ++e) {
            int brow = m0 + mh * 16 + e + ((lane < 16) ? 0 : 8);
            size_t off = (size_t)brow * HIDN + jcol;
            float gi = acc[mh][0][e] + bsum[0];
            float gf = acc[mh][1][e] + bsum[1];
            float gg = acc[mh][2][e] + bsum[2];
            float go = acc[mh][3][e] + bsum[3];
            float cn = sigmoidf_(gf) * c[off] + sigmoidf_(gi) * tanhf(gg);
            c[off]   = cn;
            hout[off] = (bf16_t)(sigmoidf_(go) * tanhf(cn));
        }
    }
}

// ---------------- FC head + decoder input packing ----------------
// Wave-per-batch-row: lanes stride j (coalesced), 7 partial sums per lane,
// butterfly shfl reduction, lane 0 writes y and packs next decoder input.
__global__ __launch_bounds__(256)
void fc_kernel(const bf16_t* __restrict__ h2,    // [B][H]
               const float*  __restrict__ fcW,   // [7][H]
               const float*  __restrict__ fcb,   // [7]
               float* __restrict__ out,          // d_out + d*7 ; batch stride PRED*7
               float* __restrict__ xdec,         // [B][10]
               const float* __restrict__ dte_d)  // dte + d*3 ; batch stride PRED*3
{
    int lane = threadIdx.x & 31;
    int b = blockIdx.x * (blockDim.x >> 5) + (threadIdx.x >> 5);
    if (b >= BATCH) return;

    float acc[OUTW];
#pragma unroll
    for (int o = 0; o < OUTW; ++o) acc[o] = 0.f;

    const bf16_t* hr = h2 + (size_t)b * HIDN;
    for (int j = lane; j < HIDN; j += 32) {
        float hv = (float)hr[j];
#pragma unroll
        for (int o = 0; o < OUTW; ++o) acc[o] += hv * fcW[(size_t)o * HIDN + j];
    }
#pragma unroll
    for (int o = 0; o < OUTW; ++o) {
#pragma unroll
        for (int s = 16; s > 0; s >>= 1)
            acc[o] += __shfl_xor(acc[o], s, 32);
    }
    if (lane == 0) {
#pragma unroll
        for (int o = 0; o < OUTW; ++o) {
            float y = acc[o] + fcb[o];
            out[(size_t)b * (PRED * OUTW) + o] = y;
            xdec[b * 10 + o] = y;
        }
        const float* te = dte_d + (size_t)b * (PRED * 3);
        xdec[b * 10 + 7] = te[0];
        xdec[b * 10 + 8] = te[1];
        xdec[b * 10 + 9] = te[2];
    }
}

// ---------------- host orchestration ----------------

extern "C" void kernel_launch(void* const* d_in, const int* in_sizes, int n_in,
                              void* d_out, int out_size, void* d_ws, size_t ws_size,
                              hipStream_t stream) {
    const float* x   = (const float*)d_in[0];
    const int*   xte = (const int*)d_in[2];
    const int*   yte = (const int*)d_in[3];
    const float* e0  = (const float*)d_in[4];
    const float* e1  = (const float*)d_in[5];
    const float* e2  = (const float*)d_in[6];
    const float* e3  = (const float*)d_in[7];
    const float* Wih[3] = {(const float*)d_in[8],  (const float*)d_in[12], (const float*)d_in[16]};
    const float* Whh[3] = {(const float*)d_in[9],  (const float*)d_in[13], (const float*)d_in[17]};
    const float* bih[3] = {(const float*)d_in[10], (const float*)d_in[14], (const float*)d_in[18]};
    const float* bhh[3] = {(const float*)d_in[11], (const float*)d_in[15], (const float*)d_in[19]};
    const float* fcW = (const float*)d_in[20];
    const float* fcb = (const float*)d_in[21];
    float* out = (float*)d_out;

    // workspace carving (all sizes multiples of 256B; hbuf & cbuf contiguous for one zero-fill)
    size_t off = 0;
    auto carve = [&](size_t bytes) -> void* {
        void* p = (char*)d_ws + off;
        off += (bytes + 255) & ~(size_t)255;
        return p;
    };
    const size_t BH = (size_t)BATCH * HIDN;
    bf16_t* hbuf = (bf16_t*)carve(2 * 3 * BH * sizeof(bf16_t));   // [parity][layer][B][H]
    float*  cbuf = (float*) carve(3 * BH * sizeof(float));        // [layer][B][H]
    bf16_t* WhhB[3]; bf16_t* WihB[2];
    for (int l = 0; l < 3; ++l) WhhB[l] = (bf16_t*)carve((size_t)4 * HIDN * HIDN * sizeof(bf16_t));
    for (int l = 0; l < 2; ++l) WihB[l] = (bf16_t*)carve((size_t)4 * HIDN * HIDN * sizeof(bf16_t));
    float* xin  = (float*)carve((size_t)BATCH * TENC * 10 * sizeof(float));
    float* dte  = (float*)carve((size_t)BATCH * PRED * 3  * sizeof(float));
    float* xdec = (float*)carve((size_t)BATCH * 10 * sizeof(float));

    // zero h (both parities) and c in one fill (contiguous)
    {
        int nu32 = (int)((2 * 3 * BH * sizeof(bf16_t) + 3 * BH * sizeof(float)) / 4);
        fill_u32_kernel<<<(nu32 + 255) / 256, 256, 0, stream>>>((uint32_t*)hbuf, 0u, nu32);
    }
    // weight conversion fp32 -> bf16
    {
        int nw = 4 * HIDN * HIDN;
        for (int l = 0; l < 3; ++l)
            cvt_bf16_kernel<<<(nw + 255) / 256, 256, 0, stream>>>(Whh[l], WhhB[l], nw);
        cvt_bf16_kernel<<<(nw + 255) / 256, 256, 0, stream>>>(Wih[1], WihB[0], nw);
        cvt_bf16_kernel<<<(nw + 255) / 256, 256, 0, stream>>>(Wih[2], WihB[1], nw);
    }
    build_xin_kernel<<<(BATCH * TENC + 127) / 128, 128, 0, stream>>>(x, xte, e0, e1, e2, e3, xin);
    build_dte_kernel<<<(BATCH * PRED + 127) / 128, 128, 0, stream>>>(yte, e0, e1, e2, e3, dte);

    auto hb = [&](int l, int par) -> bf16_t* { return hbuf + ((size_t)par * 3 + l) * BH; };

    // ---- encoder: t = 0..127 ----
    for (int t = 0; t < TENC; ++t) {
        int p = t & 1, q = p ^ 1;
        lstm_gate_kernel<<<64, 256, 0, stream>>>(
            hb(0, q), WhhB[0], nullptr, nullptr,
            xin + (size_t)t * 10, TENC * 10, Wih[0],
            bih[0], bhh[0], cbuf + 0 * BH, hb(0, p));
        lstm_gate_kernel<<<64, 256, 0, stream>>>(
            hb(1, q), WhhB[1], hb(0, p), WihB[0],
            nullptr, 0, nullptr,
            bih[1], bhh[1], cbuf + 1 * BH, hb(1, p));
        lstm_gate_kernel<<<64, 256, 0, stream>>>(
            hb(2, q), WhhB[2], hb(1, p), WihB[1],
            nullptr, 0, nullptr,
            bih[2], bhh[2], cbuf + 2 * BH, hb(2, p));
    }

    // ---- y0 from final encoder state; prep decoder input with te[0] ----
    fc_kernel<<<32, 256, 0, stream>>>(
        hb(2, (TENC - 1) & 1), fcW, fcb, out + 0 * OUTW, xdec, dte + 0 * 3);

    // ---- autoregressive decode: outputs d = 1..95 ----
    for (int d = 1; d < PRED; ++d) {
        int t = TENC + d - 1;
        int p = t & 1, q = p ^ 1;
        lstm_gate_kernel<<<64, 256, 0, stream>>>(
            hb(0, q), WhhB[0], nullptr, nullptr,
            xdec, 10, Wih[0],
            bih[0], bhh[0], cbuf + 0 * BH, hb(0, p));
        lstm_gate_kernel<<<64, 256, 0, stream>>>(
            hb(1, q), WhhB[1], hb(0, p), WihB[0],
            nullptr, 0, nullptr,
            bih[1], bhh[1], cbuf + 1 * BH, hb(1, p));
        lstm_gate_kernel<<<64, 256, 0, stream>>>(
            hb(2, q), WhhB[2], hb(1, p), WihB[1],
            nullptr, 0, nullptr,
            bih[2], bhh[2], cbuf + 2 * BH, hb(2, p));
        // write output d and pack next decoder input (te index d; harmless extra write at d=95)
        fc_kernel<<<32, 256, 0, stream>>>(
            hb(2, p), fcW, fcb, out + (size_t)d * OUTW, xdec, dte + (size_t)d * 3);
    }
}
